// Attention_12077448036605
// MI455X (gfx1250) — compile-verified, matrix operand order checked by reading, not run on hardware
//
#include <hip/hip_runtime.h>
#include <hip/hip_bf16.h>
#include <math.h>

// Problem constants (match reference)
#define B_  4
#define T_  2048
#define D_  2048
#define H_  16
#define HD_ 128

typedef __attribute__((ext_vector_type(16))) __bf16 v16bf;
typedef __attribute__((ext_vector_type(8)))  float  v8f;
typedef int vi4 __attribute__((vector_size(16)));   // matches builtin's pointee type
typedef __attribute__((address_space(1))) vi4 gvi4;
typedef __attribute__((address_space(3))) vi4 lvi4;

#ifndef __has_builtin
#define __has_builtin(x) 0
#endif
#if __has_builtin(__builtin_amdgcn_global_load_async_to_lds_b128) && \
    __has_builtin(__builtin_amdgcn_s_wait_asynccnt)
#define USE_ASYNC 1
#else
#define USE_ASYNC 0
#endif

// ---------- helpers ----------
__device__ __forceinline__ unsigned short f2bf(float f) {
  unsigned int x = __float_as_uint(f);
  unsigned int r = x + 0x7FFFu + ((x >> 16) & 1u);
  return (unsigned short)(r >> 16);
}

__device__ __forceinline__ v16bf frag16(uint4 lo, uint4 hi) {
  union { uint4 q[2]; v16bf v; } u;
  u.q[0] = lo; u.q[1] = hi;
  return u.v;
}

__device__ __forceinline__ v8f zero8() {
  v8f z;
#pragma unroll
  for (int i = 0; i < 8; ++i) z[i] = 0.0f;
  return z;
}

// 16B global -> LDS copy; async (ASYNCcnt / s_wait_asynccnt) when available.
__device__ __forceinline__ void cp_b128(void* lds_dst, const void* gsrc) {
#if USE_ASYNC
  __builtin_amdgcn_global_load_async_to_lds_b128(
      (gvi4*)gsrc, (lvi4*)lds_dst, 0, 0);
#else
  *(uint4*)lds_dst = *(const uint4*)gsrc;
#endif
}
__device__ __forceinline__ void cp_wait() {
#if USE_ASYNC
  __builtin_amdgcn_s_wait_asynccnt(0);
#endif
}

// ---------- fp32 -> bf16 conversion ----------
__global__ void f32_to_bf16(const float* __restrict__ in,
                            unsigned short* __restrict__ out, int n) {
  int i = blockIdx.x * blockDim.x + threadIdx.x;
  if (i < n) out[i] = f2bf(in[i]);
}

// ---------- TN GEMM: C[M,N](f32) = A[M,K](bf16 rowmajor) * B[N,K](bf16 rowmajor)^T ----------
// block tile 128(M) x 128(N), 8 waves, wave tile 32x64 (2x4 WMMA), K step 64.
__global__ __launch_bounds__(256) void gemm_tn_bf16(
    const unsigned short* __restrict__ A, const unsigned short* __restrict__ Bw,
    float* __restrict__ C, int M, int N) {
  const int K = D_;
  __shared__ unsigned short As[128][80];  // 64 K-els + pad to 80 (160B rows, 16B aligned)
  __shared__ unsigned short Bs[128][80];

  const int tid  = threadIdx.x;
  const int lane = tid & 31;
  const int wave = tid >> 5;
  const int m0 = blockIdx.y * 128;
  const int n0 = blockIdx.x * 128;
  const int wM = (wave >> 1) * 32;  // 0,32,64,96
  const int wN = (wave & 1) * 64;   // 0,64
  const int lrow = lane & 15;
  const int hl   = lane >> 4;
  const int kb   = hl * 8;

  v8f acc[2][4];
#pragma unroll
  for (int i = 0; i < 2; ++i)
#pragma unroll
    for (int j = 0; j < 4; ++j) acc[i][j] = zero8();

  for (int k0 = 0; k0 < K; k0 += 64) {
    // load A,B tiles 128x64 bf16 (1024 uint4 each, 4 per thread each)
#pragma unroll
    for (int r = 0; r < 4; ++r) {
      int idx = tid + 256 * r;
      int row = idx >> 3;  // 8 uint4 per 64-el row
      int c4  = idx & 7;
      cp_b128(&As[row][c4 * 8], A  + (size_t)(m0 + row) * K + k0 + c4 * 8);
      cp_b128(&Bs[row][c4 * 8], Bw + (size_t)(n0 + row) * K + k0 + c4 * 8);
    }
    if (k0 + 64 < K)
      __builtin_prefetch(A + (size_t)(m0 + (tid >> 1)) * K + k0 + 64, 0, 3);
    cp_wait();
    __syncthreads();

#pragma unroll
    for (int c = 0; c < 2; ++c) {
      v16bf af[2], bfr[4];
#pragma unroll
      for (int mi = 0; mi < 2; ++mi) {
        int row = wM + mi * 16 + lrow;
        af[mi] = frag16(*(const uint4*)&As[row][32 * c + kb],
                        *(const uint4*)&As[row][32 * c + kb + 16]);
      }
#pragma unroll
      for (int ni = 0; ni < 4; ++ni) {
        int row = wN + ni * 16 + lrow;
        bfr[ni] = frag16(*(const uint4*)&Bs[row][32 * c + kb],
                         *(const uint4*)&Bs[row][32 * c + kb + 16]);
      }
#pragma unroll
      for (int mi = 0; mi < 2; ++mi)
#pragma unroll
        for (int ni = 0; ni < 4; ++ni)
          acc[mi][ni] = __builtin_amdgcn_wmma_f32_16x16x32_bf16(
              false, af[mi], false, bfr[ni], (short)0, acc[mi][ni], false, false);
    }
    __syncthreads();
  }

  // store: C-frag layout: VGPR r -> M = r + hl*8 ; N = lane%16
#pragma unroll
  for (int mi = 0; mi < 2; ++mi)
#pragma unroll
    for (int ni = 0; ni < 4; ++ni)
#pragma unroll
      for (int r = 0; r < 8; ++r) {
        int mm = m0 + wM + mi * 16 + hl * 8 + r;
        int nn = n0 + wN + ni * 16 + lrow;
        C[(size_t)mm * N + nn] = acc[mi][ni][r];
      }
}

// ---------- RoPE on Q,K (fp32 in), writes bf16 in (B,H,T,HD) layout; scales Q by 1/sqrt(HD) ----------
__global__ void rope_convert(const float* __restrict__ qf, const float* __restrict__ kf,
                             const float* __restrict__ fcos, const float* __restrict__ fsin,
                             unsigned short* __restrict__ qb, unsigned short* __restrict__ kb) {
  long idx = (long)blockIdx.x * blockDim.x + threadIdx.x;  // B*H*T*(HD/2)
  int  i   = (int)(idx & 63);
  long rest = idx >> 6;
  int  t = (int)(rest % T_);
  long bh = rest / T_;
  int  h = (int)(bh % H_);
  int  b = (int)(bh / H_);
  float c = fcos[t * 64 + i], s = fsin[t * 64 + i];
  size_t gsrc = ((size_t)b * T_ + t) * D_ + h * HD_ + 2 * i;
  float qr = qf[gsrc], qi = qf[gsrc + 1];
  float kr = kf[gsrc], ki = kf[gsrc + 1];
  const float scale = 0.08838834764831845f;  // 1/sqrt(128) folded into Q
  size_t gdst = ((size_t)bh * T_ + t) * HD_ + 2 * i;
  qb[gdst]     = f2bf((qr * c - qi * s) * scale);
  qb[gdst + 1] = f2bf((qr * s + qi * c) * scale);
  kb[gdst]     = f2bf(kr * c - ki * s);
  kb[gdst + 1] = f2bf(kr * s + ki * c);
}

// ---------- V fp32 (B,T,D) -> bf16 (B,H,T,HD) ----------
__global__ void v_convert(const float* __restrict__ vf, unsigned short* __restrict__ vb) {
  long idx = (long)blockIdx.x * blockDim.x + threadIdx.x;  // B*T*D
  int d = (int)(idx % D_);
  long rest = idx / D_;
  int t = (int)(rest % T_);
  int b = (int)(rest / T_);
  int h = d / HD_, hd = d % HD_;
  vb[(((size_t)b * H_ + h) * T_ + t) * HD_ + hd] = f2bf(vf[idx]);
}

// ---------- flash attention: per block = (b,h) x 128 query rows; 8 waves x 16 rows ----------
__global__ __launch_bounds__(256) void flash_attn(
    const unsigned short* __restrict__ Q, const unsigned short* __restrict__ Kg,
    const unsigned short* __restrict__ Vg, unsigned short* __restrict__ Og) {
  __shared__ unsigned short Ks[64][144];     // 64 keys x 128 hd (288B rows)
  __shared__ unsigned short Vt[128][80];     // transposed: hd x t (160B rows)
  __shared__ unsigned short Ps[8][16][80];   // per-wave P staging (C-layout -> A-layout)

  const int tid  = threadIdx.x;
  const int lane = tid & 31;
  const int wave = tid >> 5;
  const int lrow = lane & 15;
  const int hl   = lane >> 4;
  const int kb   = hl * 8;

  const int bh = blockIdx.x;           // b*H + h
  const int q0 = blockIdx.y * 128;
  const int b  = bh / H_;
  const int h  = bh % H_;
  const size_t base = (size_t)bh * T_ * HD_;

  // Q fragments: 16 rows x 128 hd = 4 chunks of 16x32 (A layout)
  v16bf qfrag[4];
  {
    const unsigned short* qrow = Q + base + (size_t)(q0 + wave * 16 + lrow) * HD_;
#pragma unroll
    for (int c = 0; c < 4; ++c)
      qfrag[c] = frag16(*(const uint4*)(qrow + 32 * c + kb),
                        *(const uint4*)(qrow + 32 * c + kb + 16));
  }

  v8f o[8];
  float mrow[8], lsum[8];
#pragma unroll
  for (int n = 0; n < 8; ++n) o[n] = zero8();
#pragma unroll
  for (int r = 0; r < 8; ++r) { mrow[r] = -INFINITY; lsum[r] = 0.0f; }

  const int kend = q0 + 128;   // causal: only key tiles up to the query tile
  for (int k0 = 0; k0 < kend; k0 += 64) {
    // K tile 64x128 bf16 (1024 uint4, 4 per thread) via async copy when available
#pragma unroll
    for (int r = 0; r < 4; ++r) {
      int idx = tid + 256 * r;
      int row = idx >> 4;
      int c4  = idx & 15;
      cp_b128(&Ks[row][c4 * 8],
              Kg + base + (size_t)(k0 + row) * HD_ + c4 * 8);
    }
    // V tile transposed into Vt[hd][t] (needs element shuffle -> regular path)
#pragma unroll
    for (int r = 0; r < 4; ++r) {
      int idx = tid + 256 * r;
      int row = idx >> 4;  // t within tile
      int c4  = idx & 15;  // hd/8
      uint4 d = *(const uint4*)(Vg + base + (size_t)(k0 + row) * HD_ + c4 * 8);
      const unsigned short* e = (const unsigned short*)&d;
#pragma unroll
      for (int j = 0; j < 8; ++j) Vt[c4 * 8 + j][row] = e[j];
    }
    cp_wait();
    __syncthreads();

    // S = Q * K^T for 4 key sub-tiles of 16 (K-dim = HD = 4 WMMAs each)
    v8f sfr[4];
#pragma unroll
    for (int nt = 0; nt < 4; ++nt) {
      v8f s = zero8();
      int krow = nt * 16 + lrow;
#pragma unroll
      for (int c = 0; c < 4; ++c) {
        v16bf bfr = frag16(*(const uint4*)&Ks[krow][32 * c + kb],
                           *(const uint4*)&Ks[krow][32 * c + kb + 16]);
        s = __builtin_amdgcn_wmma_f32_16x16x32_bf16(
            false, qfrag[c], false, bfr, (short)0, s, false, false);
      }
      sfr[nt] = s;
    }

    // causal mask: element r -> query row q0+wave*16+hl*8+r ; key k0+nt*16+lrow
    const int tqb = q0 + wave * 16 + hl * 8;
#pragma unroll
    for (int nt = 0; nt < 4; ++nt)
#pragma unroll
      for (int r = 0; r < 8; ++r)
        if (k0 + nt * 16 + lrow > tqb + r) sfr[nt][r] = -INFINITY;

    // online softmax: row stats reduced across 16 lanes within a half
    float mnew[8], alpha[8], rsum[8];
#pragma unroll
    for (int r = 0; r < 8; ++r) {
      float mx = mrow[r];
#pragma unroll
      for (int nt = 0; nt < 4; ++nt) mx = fmaxf(mx, sfr[nt][r]);
#pragma unroll
      for (int d = 1; d < 16; d <<= 1) mx = fmaxf(mx, __shfl_xor(mx, d, 32));
      mnew[r]  = mx;
      alpha[r] = __expf(mrow[r] - mx);
      mrow[r]  = mx;
      rsum[r]  = 0.0f;
    }
#pragma unroll
    for (int nt = 0; nt < 4; ++nt)
#pragma unroll
      for (int r = 0; r < 8; ++r) {
        float p = __expf(sfr[nt][r] - mnew[r]);
        sfr[nt][r] = p;
        rsum[r] += p;
      }
#pragma unroll
    for (int r = 0; r < 8; ++r) {
#pragma unroll
      for (int d = 1; d < 16; d <<= 1) rsum[r] += __shfl_xor(rsum[r], d, 32);
      lsum[r] = lsum[r] * alpha[r] + rsum[r];
    }
#pragma unroll
    for (int n = 0; n < 8; ++n)
#pragma unroll
      for (int r = 0; r < 8; ++r) o[n][r] *= alpha[r];

    // stage P (C-layout) into LDS, reload in A-layout for the PV WMMA
#pragma unroll
    for (int nt = 0; nt < 4; ++nt)
#pragma unroll
      for (int r = 0; r < 8; ++r)
        Ps[wave][hl * 8 + r][nt * 16 + lrow] = f2bf(sfr[nt][r]);
    v16bf pf[2];
#pragma unroll
    for (int c = 0; c < 2; ++c)
      pf[c] = frag16(*(const uint4*)&Ps[wave][lrow][32 * c + kb],
                     *(const uint4*)&Ps[wave][lrow][32 * c + kb + 16]);

    // O += P * V   (8 hd sub-tiles, K-dim = 64 keys = 2 WMMAs each)
#pragma unroll
    for (int nt2 = 0; nt2 < 8; ++nt2) {
      int vrow = nt2 * 16 + lrow;
#pragma unroll
      for (int c = 0; c < 2; ++c) {
        v16bf vfr = frag16(*(const uint4*)&Vt[vrow][32 * c + kb],
                           *(const uint4*)&Vt[vrow][32 * c + kb + 16]);
        o[nt2] = __builtin_amdgcn_wmma_f32_16x16x32_bf16(
            false, pf[c], false, vfr, (short)0, o[nt2], false, false);
      }
    }
    __syncthreads();
  }

  // epilogue: divide by l, write bf16 to (B,T,D) layout for final projection
#pragma unroll
  for (int r = 0; r < 8; ++r) {
    float inv = 1.0f / lsum[r];
    int tq = q0 + wave * 16 + hl * 8 + r;
#pragma unroll
    for (int nt2 = 0; nt2 < 8; ++nt2) {
      int hd = nt2 * 16 + lrow;
      Og[((size_t)b * T_ + tq) * D_ + h * HD_ + hd] = f2bf(o[nt2][r] * inv);
    }
  }
}

// ---------- host ----------
extern "C" void kernel_launch(void* const* d_in, const int* in_sizes, int n_in,
                              void* d_out, int out_size, void* d_ws, size_t ws_size,
                              hipStream_t stream) {
  const float* x    = (const float*)d_in[0];
  const float* wq   = (const float*)d_in[1];
  const float* wk   = (const float*)d_in[2];
  const float* wv   = (const float*)d_in[3];
  const float* wo   = (const float*)d_in[4];
  const float* fcos = (const float*)d_in[5];
  const float* fsin = (const float*)d_in[6];
  // d_in[7] = mask (causality handled arithmetically)
  float* out = (float*)d_out;

  const size_t ND = (size_t)B_ * T_ * D_;   // 16,777,216
  const size_t NW = (size_t)D_ * D_;        //  4,194,304

  char* ws = (char*)d_ws;
  size_t off = 0;
  auto alloc = [&](size_t bytes) -> void* {
    void* p = ws + off;
    off += (bytes + 255) & ~(size_t)255;
    return p;
  };
  unsigned short* xb  = (unsigned short*)alloc(ND * 2);
  unsigned short* wqb = (unsigned short*)alloc(NW * 2);
  unsigned short* wkb = (unsigned short*)alloc(NW * 2);
  unsigned short* wvb = (unsigned short*)alloc(NW * 2);
  unsigned short* wob = (unsigned short*)alloc(NW * 2);
  float* qf = (float*)alloc(ND * 4);
  float* kf = (float*)alloc(ND * 4);
  float* vf = (float*)alloc(ND * 4);
  unsigned short* qb = (unsigned short*)alloc(ND * 2);
  unsigned short* kb = (unsigned short*)alloc(ND * 2);
  unsigned short* vb = (unsigned short*)alloc(ND * 2);
  unsigned short* ab = (unsigned short*)qf;  // alias: qf dead after rope

  const int thr = 256;
  // 1) convert activations + weights to bf16
  f32_to_bf16<<<(int)((ND + thr - 1) / thr), thr, 0, stream>>>(x, xb, (int)ND);
  f32_to_bf16<<<(int)((NW + thr - 1) / thr), thr, 0, stream>>>(wq, wqb, (int)NW);
  f32_to_bf16<<<(int)((NW + thr - 1) / thr), thr, 0, stream>>>(wk, wkb, (int)NW);
  f32_to_bf16<<<(int)((NW + thr - 1) / thr), thr, 0, stream>>>(wv, wvb, (int)NW);
  f32_to_bf16<<<(int)((NW + thr - 1) / thr), thr, 0, stream>>>(wo, wob, (int)NW);

  // 2) QKV projections (TN GEMM, fp32 out)
  dim3 ggrid(D_ / 128, (B_ * T_) / 128);
  gemm_tn_bf16<<<ggrid, thr, 0, stream>>>(xb, wqb, qf, B_ * T_, D_);
  gemm_tn_bf16<<<ggrid, thr, 0, stream>>>(xb, wkb, kf, B_ * T_, D_);
  gemm_tn_bf16<<<ggrid, thr, 0, stream>>>(xb, wvb, vf, B_ * T_, D_);

  // 3) RoPE (q,k) + layout change to (B,H,T,HD) bf16; V convert/relayout
  long nrope = (long)B_ * H_ * T_ * (HD_ / 2);
  rope_convert<<<(int)(nrope / thr), thr, 0, stream>>>(qf, kf, fcos, fsin, qb, kb);
  v_convert<<<(int)(ND / thr), thr, 0, stream>>>(vf, vb);

  // 4) flash attention -> bf16 (B,T,D)
  flash_attn<<<dim3(B_ * H_, T_ / 128), thr, 0, stream>>>(qb, kb, vb, ab);

  // 5) output projection -> fp32 d_out
  gemm_tn_bf16<<<ggrid, thr, 0, stream>>>(ab, wob, out, B_ * T_, D_);
  (void)in_sizes; (void)n_in; (void)out_size; (void)ws_size;
}
